// GPSRegressor_37409165149078
// MI455X (gfx1250) — matrix operand order repaired and checked by reading, not compiled, not run on hardware
//
#include <hip/hip_runtime.h>
#include <hip/hip_bf16.h>

typedef __attribute__((ext_vector_type(16))) _Float16 v16h;
typedef __attribute__((ext_vector_type(8)))  _Float16 v8h;
typedef __attribute__((ext_vector_type(8)))  float    v8f;

#define NGRAPH 128
#define NPG    256
#define CH     128
#define NL     3
#define NHEAD  4
#define HDIM   32
#define INCH   30
#define WALK   20
#define EIN    11
#define DEGF   8
#define NNODE  (NGRAPH*NPG)     // 32768
#define NEDGE  (NNODE*DEGF)     // 262144
#define BN_EPS 1e-5f

extern __shared__ char dynsmem[];

__device__ __forceinline__ v8f wmma16(v16h a, v16h b, v8f c) {
  return __builtin_amdgcn_wmma_f32_16x16x32_f16(false, a, false, b, (short)0, c,
                                                false, false);
}

// ---- WMMA operand loaders (CDNA5 ISA 7.12.2 layouts, wave32) ----
// A 16x32 f16 tile: lanes0-15 K={0..7,16..23}, lanes16-31 K={8..15,24..31};
// per lane this is two contiguous 8-element runs -> b128 loads.
template <typename T>
__device__ __forceinline__ v16h load_a16(const T* A, int lda, int row0, int k0, int lane) {
  const int half = lane >> 4;
  const T* Ar = A + (size_t)(row0 + (lane & 15)) * lda + k0;
  v16h a;
#pragma unroll
  for (int j = 0; j < 8; ++j) {
    int base = (j < 4) ? (half * 8 + 2 * j) : (16 + half * 8 + 2 * (j - 4));
    a[2 * j]     = (_Float16)Ar[base];
    a[2 * j + 1] = (_Float16)Ar[base + 1];
  }
  return a;
}

// f32-source A loader, fast path (no K guard) with optional residual addend
__device__ __forceinline__ v16h load_a32f(const float* A, const float* A2, int lda,
                                          int row0, int k0, int lane) {
  const int half = lane >> 4;
  const size_t roff = (size_t)(row0 + (lane & 15)) * lda;
  const float* Ar = A + roff + k0;
  const float* A2r = A2 ? (A2 + roff + k0) : nullptr;
  v16h a;
#pragma unroll
  for (int j = 0; j < 8; ++j) {
    int base = (j < 4) ? (half * 8 + 2 * j) : (16 + half * 8 + 2 * (j - 4));
#pragma unroll
    for (int p = 0; p < 2; ++p) {
      float v = Ar[base + p];
      if (A2r) v += A2r[base + p];
      a[2 * j + p] = (_Float16)v;
    }
  }
  return a;
}

// f32-source A loader with K guard (ragged tail chunks only)
__device__ __forceinline__ v16h load_a32(const float* A, const float* A2, int lda,
                                         int row0, int k0, int Kreal, int lane) {
  const int half = lane >> 4;
  const size_t roff = (size_t)(row0 + (lane & 15)) * lda;
  const float* Ar = A + roff;
  const float* A2r = A2 ? (A2 + roff) : nullptr;
  v16h a;
#pragma unroll
  for (int j = 0; j < 8; ++j) {
    int base = (j < 4) ? (half * 8 + 2 * j) : (16 + half * 8 + 2 * (j - 4));
#pragma unroll
    for (int p = 0; p < 2; ++p) {
      int k = k0 + base + p;
      float v = 0.f;
      if (k < Kreal) { v = Ar[k]; if (A2r) v += A2r[k]; }
      a[2 * j + p] = (_Float16)v;
    }
  }
  return a;
}

// B loader from TRANSPOSED storage: logical B[k][n] = Bt[n*ld + k].
// Per lane: two 8-element runs K={8h..8h+7, 16+8h..16+8h+7} -> b128-vectorizable. T = f16 or f32.
template <typename T>
__device__ __forceinline__ v16h load_bT(const T* Bt, int ld, int k0, int n0, int lane) {
  const int n = n0 + (lane & 15);
  const int kb = k0 + ((lane >> 4) << 3);
  const T* r = Bt + (size_t)n * ld + kb;
  v16h b;
#pragma unroll
  for (int v = 0; v < 8; ++v) {
    int o = 2 * v + ((v >= 4) ? 8 : 0);
    b[2 * v]     = (_Float16)r[o];
    b[2 * v + 1] = (_Float16)r[o + 1];
  }
  return b;
}

// ======================= utility kernels =======================
__global__ void k_zero(float* p, int n) {
  int i = blockIdx.x * blockDim.x + threadIdx.x;
  if (i < n) p[i] = 0.f;
}

__global__ void k_scatterA(const int* ei, float* A) {
  int e = blockIdx.x * blockDim.x + threadIdx.x;
  if (e >= NEDGE) return;
  int s = ei[e], d = ei[NEDGE + e];
  int g = s / NPG;
  int sl = s - g * NPG, dl = d - g * NPG;
  atomicAdd(&A[(size_t)g * NPG * NPG + sl * NPG + dl], 1.f);
}

__global__ void k_deg(const float* A, float* deg) {
  int gid = blockIdx.x * blockDim.x + threadIdx.x;
  if (gid >= NGRAPH * NPG) return;
  int b = gid >> 8, r = gid & 255;
  const float* row = A + (size_t)b * NPG * NPG + (size_t)r * NPG;
  float s = 1.f;  // self loop
  for (int c = 0; c < NPG; ++c) s += row[c];
  deg[gid] = fmaxf(s, 1.f);
}

// M[b][i][j] = (A[b][j][i] + I) / deg[b][j]
__global__ void k_buildM(const float* A, const float* deg, _Float16* M) {
  int gid = blockIdx.x * blockDim.x + threadIdx.x;
  if (gid >= NGRAPH * NPG * NPG) return;
  int b = gid >> 16, rem = gid & 65535;
  int i = rem >> 8, j = rem & 255;
  float a = A[(size_t)b * NPG * NPG + (size_t)j * NPG + i] + (i == j ? 1.f : 0.f);
  M[gid] = (_Float16)(a / deg[b * NPG + j]);
}

// Transpose + f16-convert a weight matrix: Wt[n][k] = W[k][n], zero-pad K->Kpad
__global__ void k_wt(const float* W, _Float16* Wt, int K, int N, int Kpad) {
  int gid = blockIdx.x * blockDim.x + threadIdx.x;
  if (gid >= N * Kpad) return;
  int n = gid / Kpad, k = gid % Kpad;
  Wt[gid] = (k < K) ? (_Float16)W[(size_t)k * N + n] : (_Float16)0.0f;
}

// ---- RWSE: per-graph P <- M P (P kept TRANSPOSED), diag -> pe.
// M LDS-resident (128KB) via async-to-LDS DMA; 512 threads = 16 waves (4/SIMD32).
__global__ __launch_bounds__(512) void k_rwse(const _Float16* Mg, _Float16* Pa,
                                              _Float16* Pb, float* pe) {
  _Float16* Ms = (_Float16*)dynsmem;  // 256*256 f16
  int g = blockIdx.x, tid = threadIdx.x;
  // async DMA M tile into LDS (each lane: 16B, independent LDS dest address)
  {
    const char* gsrc = (const char*)(Mg + (size_t)g * NPG * NPG);
    char* lbase = (char*)Ms;
#pragma unroll 4
    for (int j = 0; j < 16; ++j) {
      int off = (tid + j * 512) * 16;
      unsigned lds = (unsigned)(size_t)(void*)(lbase + off);
      unsigned long long ga = (unsigned long long)(size_t)(gsrc + off);
      asm volatile("global_load_async_to_lds_b128 %0, %1, off"
                   :: "v"(lds), "v"(ga) : "memory");
    }
  }
  // init P^T = I (identity is symmetric)
  _Float16* Pin = Pa + (size_t)g * NPG * NPG;
  _Float16* Pout = Pb + (size_t)g * NPG * NPG;
  for (int i = tid; i < NPG * NPG; i += 512) {
    int r = i >> 8, c = i & 255;
    Pin[i] = (r == c) ? (_Float16)1.0f : (_Float16)0.0f;
  }
  asm volatile("s_wait_asynccnt 0x0" ::: "memory");
  __threadfence();
  __syncthreads();
  int wave = tid >> 5, lane = tid & 31;
  int half = lane >> 4, l15 = lane & 15;
  for (int step = 0; step < WALK; ++step) {
    for (int t = wave; t < 256; t += 16) {  // 16x16 grid of 16x16 tiles
      int tm = t >> 4, tn = t & 15;
      v8f acc = {};
#pragma unroll
      for (int k0 = 0; k0 < NPG; k0 += 32) {
        v16h a = load_a16<_Float16>(Ms, NPG, tm * 16, k0, lane);   // ds b128
        v16h b = load_bT<_Float16>(Pin, NPG, k0, tn * 16, lane);   // global b128
        acc = wmma16(a, b, acc);
      }
      int n = tn * 16 + l15;
      int rb = tm * 16 + half * 8;           // 8 consecutive output rows
      v8h hv;
#pragma unroll
      for (int i = 0; i < 8; ++i) hv[i] = (_Float16)acc[i];
      *(v8h*)(Pout + (size_t)n * NPG + rb) = hv;   // transposed, one b128 store
    }
    __threadfence();
    __syncthreads();
    // diag pass, out of the hot loop: one lane per node (diag invariant under T)
    if (tid < NPG)
      pe[(size_t)(g * NPG + tid) * WALK + step] = (float)Pout[(size_t)tid * NPG + tid];
    _Float16* tmp = Pin; Pin = Pout; Pout = tmp;
    // no extra barrier: next step only READS new Pin; writes go to the other buffer
  }
}

__global__ void k_concat(const float* x, const float* pe, float* cat) {
  int gid = blockIdx.x * blockDim.x + threadIdx.x;
  if (gid >= NNODE * (INCH + WALK)) return;
  int n = gid / (INCH + WALK), j = gid % (INCH + WALK);
  cat[gid] = (j < INCH) ? x[(size_t)n * INCH + j] : pe[(size_t)n * WALK + (j - INCH)];
}

// ---- generic WMMA GEMM: C = act( (A[+A2]) * W + bias ), W pre-transposed f16 ----
__global__ __launch_bounds__(256) void k_gemm(const float* A, const float* A2, int lda,
                                              const _Float16* Bt, const float* bias,
                                              float* C, int ldc, int Mtiles, int Ntiles,
                                              int Kreal, int Kpad, int relu) {
  int wave = threadIdx.x >> 5, lane = threadIdx.x & 31;
  int gw = blockIdx.x * 8 + wave;
  if (gw >= Mtiles * Ntiles) return;
  int tm = gw / Ntiles, tn = gw % Ntiles;
  __builtin_prefetch(A + (size_t)(tm * 16 + (lane & 15)) * lda, 0, 1);
  v8f acc = {};
  for (int k0 = 0; k0 < Kpad; k0 += 32) {
    v16h a = (k0 + 32 <= Kreal)
                 ? load_a32f(A, A2, lda, tm * 16, k0, lane)
                 : load_a32(A, A2, lda, tm * 16, k0, Kreal, lane);
    v16h b = load_bT<_Float16>(Bt, Kpad, k0, tn * 16, lane);
    acc = wmma16(a, b, acc);
  }
  int half = lane >> 4, l15 = lane & 15;
  int n = tn * 16 + l15;
  float bv = bias ? bias[n] : 0.f;
#pragma unroll
  for (int i = 0; i < 8; ++i) {
    float v = acc[i] + bv;
    if (relu) v = fmaxf(v, 0.f);
    C[(size_t)(tm * 16 + half * 8 + i) * ldc + n] = v;
  }
}

// Wc[l] = ep_W @ ge_W[l] (11x128), bc[l] = ep_b @ ge_W[l] + ge_b[l]
__global__ void k_wc(const float* epW, const float* epb, const float* geW,
                     const float* geb, float* Wc, float* bc) {
  int gid = blockIdx.x * blockDim.x + threadIdx.x;
  if (gid >= NL * (EIN + 1) * CH) return;
  int l = gid / ((EIN + 1) * CH);
  int rem = gid % ((EIN + 1) * CH);
  int row = rem / CH, c = rem % CH;
  const float* G = geW + (size_t)l * CH * CH;
  if (row < EIN) {
    float acc = 0.f;
    for (int m = 0; m < CH; ++m) acc += epW[row * CH + m] * G[m * CH + c];
    Wc[(size_t)l * EIN * CH + row * CH + c] = acc;
  } else {
    float acc = geb[l * CH + c];
    for (int m = 0; m < CH; ++m) acc += epb[m] * G[m * CH + c];
    bc[l * CH + c] = acc;
  }
}

// GINE message: aggr[dst] += relu(h[src] + edge_attr @ Wc + bc), 4 channels/thread
__global__ void k_msg(const int* ei, const float* ea, const float* h,
                      const float* Wc, const float* bc, float* aggr) {
  int gid = blockIdx.x * blockDim.x + threadIdx.x;
  if (gid >= NEDGE * 32) return;
  int e = gid >> 5, c4 = (gid & 31) * 4;
  int s = ei[e], d = ei[NEDGE + e];
  float ev[EIN];
  const float* eav = ea + (size_t)e * EIN;
#pragma unroll
  for (int j = 0; j < EIN; ++j) ev[j] = eav[j];
  const float* hs = h + (size_t)s * CH;
#pragma unroll
  for (int q = 0; q < 4; ++q) {
    int c = c4 + q;
    float acc = bc[c];
#pragma unroll
    for (int j = 0; j < EIN; ++j) acc += ev[j] * Wc[j * CH + c];
    float m = fmaxf(hs[c] + acc, 0.f);
    atomicAdd(&aggr[(size_t)d * CH + c], m);
  }
}

// out = BN(a + b) (inference-mode batchnorm)
__global__ void k_bnadd(const float* a, const float* b, const float* g, const float* bet,
                        const float* m, const float* v, float* out) {
  int i = blockIdx.x * blockDim.x + threadIdx.x;
  if (i >= NNODE * CH) return;
  int c = i & (CH - 1);
  float x = a[i] + b[i];
  out[i] = (x - m[c]) * rsqrtf(v[c] + BN_EPS) * g[c] + bet[c];
}

__global__ void k_add(const float* a, const float* b, float* out, int n) {
  int i = blockIdx.x * blockDim.x + threadIdx.x;
  if (i < n) out[i] = a[i] + b[i];
}

// ---- attention: one block per (graph, head); S (256KB) + V^T (32KB) in LDS ----
__global__ __launch_bounds__(256) void k_attn(const float* qkv, float* oattn) {
  float* S  = (float*)dynsmem;                       // 256*256 f32
  float* VT = (float*)(dynsmem + NPG * NPG * 4);     // 32*256 f32 (transposed V)
  int bh = blockIdx.x;
  int g = bh >> 2, hh = bh & 3;
  int tid = threadIdx.x, wave = tid >> 5, lane = tid & 31;
  int half = lane >> 4, l15 = lane & 15;
  const float* base = qkv + (size_t)g * NPG * (3 * CH);
  const int qoff = hh * HDIM, koff = CH + hh * HDIM, voff = 2 * CH + hh * HDIM;
  const float scale = 0.17677669529663689f;  // 1/sqrt(32)
  // async DMA V into LDS, transposed on the fly via per-lane LDS dest addresses
  for (int idx = tid; idx < NPG * HDIM; idx += 256) {
    int node = idx >> 5, dim = idx & 31;
    unsigned lds = (unsigned)(size_t)(void*)(VT + (size_t)dim * NPG + node);
    unsigned long long ga =
        (unsigned long long)(size_t)(base + (size_t)node * (3 * CH) + voff + dim);
    asm volatile("global_load_async_to_lds_b32 %0, %1, off"
                 :: "v"(lds), "v"(ga) : "memory");
  }
  // phase 1: S = Q K^T * scale
  for (int t = wave; t < 256; t += 8) {
    int tm = t >> 4, tn = t & 15;
    v8f acc = {};
    v16h a = load_a32f(base + qoff, nullptr, 3 * CH, tm * 16, 0, lane);
    v16h b = load_bT<float>(base + koff, 3 * CH, 0, tn * 16, lane);
    acc = wmma16(a, b, acc);
    int n = tn * 16 + l15;
#pragma unroll
    for (int i = 0; i < 8; ++i)
      S[(size_t)(tm * 16 + half * 8 + i) * NPG + n] = acc[i] * scale;
  }
  __syncthreads();
  // phase 2: row softmax (thread per row)
  {
    float* row = S + (size_t)tid * NPG;
    float mx = -1e30f;
    for (int c = 0; c < NPG; ++c) mx = fmaxf(mx, row[c]);
    float sum = 0.f;
    for (int c = 0; c < NPG; ++c) { float e = __expf(row[c] - mx); row[c] = e; sum += e; }
    float inv = 1.f / sum;
    for (int c = 0; c < NPG; ++c) row[c] *= inv;
  }
  asm volatile("s_wait_asynccnt 0x0" ::: "memory");
  __syncthreads();
  // phase 3: O = S @ V  (256 x 32), V^T from LDS
  for (int t = wave; t < 32; t += 8) {
    int tm = t >> 1, tn = t & 1;
    v8f acc = {};
#pragma unroll
    for (int k0 = 0; k0 < NPG; k0 += 32) {
      v16h a = load_a16<float>(S, NPG, tm * 16, k0, lane);
      v16h b = load_bT<float>(VT, NPG, k0, tn * 16, lane);
      acc = wmma16(a, b, acc);
    }
    int n = tn * 16 + l15;
#pragma unroll
    for (int i = 0; i < 8; ++i)
      oattn[(size_t)(g * NPG + tm * 16 + half * 8 + i) * CH + hh * HDIM + n] = acc[i];
  }
}

__global__ void k_pool(const float* h, float* pooled) {
  int g = blockIdx.x, c = threadIdx.x;
  float s = 0.f;
  for (int n = 0; n < NPG; ++n) s += h[(size_t)(g * NPG + n) * CH + c];
  pooled[g * CH + c] = s * (1.f / NPG);
}

__global__ void k_head2(const float* h1o, const float* w, const float* b, float* out) {
  int g = threadIdx.x;
  if (g >= NGRAPH) return;
  float acc = b[0];
  for (int j = 0; j < 64; ++j) acc += h1o[g * 64 + j] * w[j];
  out[g] = acc;
}

// ======================= host orchestration =======================
static inline void gemm(hipStream_t s, const float* A, const float* A2, int lda,
                        const _Float16* Bt, const float* bias, float* C, int ldc,
                        int M, int N, int Kreal, int Kpad, int relu) {
  int mt = M / 16, nt = N / 16;
  int blocks = (mt * nt + 7) / 8;
  k_gemm<<<blocks, 256, 0, s>>>(A, A2, lda, Bt, bias, C, ldc, mt, nt, Kreal, Kpad, relu);
}

extern "C" void kernel_launch(void* const* d_in, const int* in_sizes, int n_in,
                              void* d_out, int out_size, void* d_ws, size_t ws_size,
                              hipStream_t stream) {
  const float* x     = (const float*)d_in[0];
  const float* ea    = (const float*)d_in[1];
  const float* in_W  = (const float*)d_in[2];
  const float* in_b  = (const float*)d_in[3];
  const float* ep_W  = (const float*)d_in[4];
  const float* ep_b  = (const float*)d_in[5];
  const float* ge_W  = (const float*)d_in[6];
  const float* ge_b  = (const float*)d_in[7];
  const float* g1_W  = (const float*)d_in[8];
  const float* g1_b  = (const float*)d_in[9];
  const float* g2_W  = (const float*)d_in[10];
  const float* g2_b  = (const float*)d_in[11];
  const float* qkv_W = (const float*)d_in[12];
  const float* qkv_b = (const float*)d_in[13];
  const float* o_W   = (const float*)d_in[14];
  const float* o_b   = (const float*)d_in[15];
  const float* bn1_g = (const float*)d_in[16];
  const float* bn1_b = (const float*)d_in[17];
  const float* bn1_m = (const float*)d_in[18];
  const float* bn1_v = (const float*)d_in[19];
  const float* bn2_g = (const float*)d_in[20];
  const float* bn2_b = (const float*)d_in[21];
  const float* bn2_m = (const float*)d_in[22];
  const float* bn2_v = (const float*)d_in[23];
  const float* bn3_g = (const float*)d_in[24];
  const float* bn3_b = (const float*)d_in[25];
  const float* bn3_m = (const float*)d_in[26];
  const float* bn3_v = (const float*)d_in[27];
  const float* m1_W  = (const float*)d_in[28];
  const float* m1_b  = (const float*)d_in[29];
  const float* m2_W  = (const float*)d_in[30];
  const float* m2_b  = (const float*)d_in[31];
  const float* h1_W  = (const float*)d_in[32];
  const float* h1_b  = (const float*)d_in[33];
  const float* h2_W  = (const float*)d_in[34];
  const float* h2_b  = (const float*)d_in[35];
  const int*   ei    = (const int*)d_in[36];
  float* out = (float*)d_out;

  char* w = (char*)d_ws;
  auto alloc = [&](size_t bytes) -> char* {
    char* p = w;
    w += (bytes + 255) & ~(size_t)255;
    return p;
  };
  float*    A     = (float*)alloc(sizeof(float) * NGRAPH * NPG * NPG);
  float*    deg   = (float*)alloc(sizeof(float) * NGRAPH * NPG);
  _Float16* M     = (_Float16*)alloc(2ull * NGRAPH * NPG * NPG);
  _Float16* P0    = (_Float16*)alloc(2ull * NGRAPH * NPG * NPG);
  _Float16* P1    = (_Float16*)alloc(2ull * NGRAPH * NPG * NPG);
  float*    pe    = (float*)alloc(sizeof(float) * NNODE * WALK);
  float*    cat   = (float*)alloc(sizeof(float) * NNODE * (INCH + WALK));
  float*    h     = (float*)alloc(sizeof(float) * NNODE * CH);
  float*    aggr  = (float*)alloc(sizeof(float) * NNODE * CH);
  float*    t1    = (float*)alloc(sizeof(float) * NNODE * CH);
  float*    t2    = (float*)alloc(sizeof(float) * NNODE * CH);
  float*    h1b   = (float*)alloc(sizeof(float) * NNODE * CH);
  float*    qkvb  = (float*)alloc(sizeof(float) * NNODE * 3 * CH);
  float*    oat   = (float*)alloc(sizeof(float) * NNODE * CH);
  float*    ob    = (float*)alloc(sizeof(float) * NNODE * CH);
  float*    h2b   = (float*)alloc(sizeof(float) * NNODE * CH);
  float*    outb  = (float*)alloc(sizeof(float) * NNODE * CH);
  float*    ff1   = (float*)alloc(sizeof(float) * NNODE * 2 * CH);
  float*    ff2   = (float*)alloc(sizeof(float) * NNODE * CH);
  float*    Wc    = (float*)alloc(sizeof(float) * NL * EIN * CH);
  float*    bc    = (float*)alloc(sizeof(float) * NL * CH);
  float*    pool  = (float*)alloc(sizeof(float) * NGRAPH * CH);
  float*    hd1   = (float*)alloc(sizeof(float) * NGRAPH * 64);
  // transposed f16 weights
  _Float16* inWt  = (_Float16*)alloc(2ull * CH * 64);
  _Float16* g1Wt  = (_Float16*)alloc(2ull * NL * CH * CH);
  _Float16* g2Wt  = (_Float16*)alloc(2ull * NL * CH * CH);
  _Float16* qkvWt = (_Float16*)alloc(2ull * NL * 3 * CH * CH);
  _Float16* oWt   = (_Float16*)alloc(2ull * NL * CH * CH);
  _Float16* m1Wt  = (_Float16*)alloc(2ull * NL * 2 * CH * CH);
  _Float16* m2Wt  = (_Float16*)alloc(2ull * NL * CH * 2 * CH);
  _Float16* h1Wt  = (_Float16*)alloc(2ull * 64 * CH);

  const int T = 256;
  const size_t cc = (size_t)CH * CH;
  // ---- pre-transpose weights to f16 ----
  k_wt<<<(CH * 64 + T - 1) / T, T, 0, stream>>>(in_W, inWt, INCH + WALK, CH, 64);
  for (int l = 0; l < NL; ++l) {
    k_wt<<<(CH * CH + T - 1) / T, T, 0, stream>>>(g1_W + l * cc, g1Wt + l * cc, CH, CH, CH);
    k_wt<<<(CH * CH + T - 1) / T, T, 0, stream>>>(g2_W + l * cc, g2Wt + l * cc, CH, CH, CH);
    k_wt<<<(3 * CH * CH + T - 1) / T, T, 0, stream>>>(qkv_W + l * 3 * cc, qkvWt + l * 3 * cc,
                                                      CH, 3 * CH, CH);
    k_wt<<<(CH * CH + T - 1) / T, T, 0, stream>>>(o_W + l * cc, oWt + l * cc, CH, CH, CH);
    k_wt<<<(2 * CH * CH + T - 1) / T, T, 0, stream>>>(m1_W + l * 2 * cc, m1Wt + l * 2 * cc,
                                                      CH, 2 * CH, CH);
    k_wt<<<(2 * CH * CH + T - 1) / T, T, 0, stream>>>(m2_W + l * 2 * cc, m2Wt + l * 2 * cc,
                                                      2 * CH, CH, 2 * CH);
  }
  k_wt<<<(64 * CH + T - 1) / T, T, 0, stream>>>(h1_W, h1Wt, CH, 64, CH);

  // ---- RWSE positional encoding ----
  {
    int nA = NGRAPH * NPG * NPG;
    k_zero<<<(nA + T - 1) / T, T, 0, stream>>>(A, nA);
    k_scatterA<<<(NEDGE + T - 1) / T, T, 0, stream>>>(ei, A);
    k_deg<<<(NGRAPH * NPG + T - 1) / T, T, 0, stream>>>(A, deg);
    k_buildM<<<(nA + T - 1) / T, T, 0, stream>>>(A, deg, M);
    k_rwse<<<NGRAPH, 512, NPG * NPG * 2, stream>>>(M, P0, P1, pe);
  }
  // ---- input projection ----
  {
    int nc = NNODE * (INCH + WALK);
    k_concat<<<(nc + T - 1) / T, T, 0, stream>>>(x, pe, cat);
    gemm(stream, cat, nullptr, INCH + WALK, inWt, in_b, h, CH,
         NNODE, CH, INCH + WALK, 64, 0);
  }
  // ---- fused edge_proj @ ge_W per layer ----
  {
    int nw = NL * (EIN + 1) * CH;
    k_wc<<<(nw + T - 1) / T, T, 0, stream>>>(ep_W, ep_b, ge_W, ge_b, Wc, bc);
  }
  // ---- GPS layers ----
  for (int l = 0; l < NL; ++l) {
    // GINE
    k_zero<<<(NNODE * CH + T - 1) / T, T, 0, stream>>>(aggr, NNODE * CH);
    k_msg<<<(NEDGE * 32 + T - 1) / T, T, 0, stream>>>(
        ei, ea, h, Wc + (size_t)l * EIN * CH, bc + l * CH, aggr);
    gemm(stream, h, aggr, CH, g1Wt + l * cc, g1_b + l * CH, t1, CH, NNODE, CH, CH, CH, 1);
    gemm(stream, t1, nullptr, CH, g2Wt + l * cc, g2_b + l * CH, t2, CH, NNODE, CH, CH, CH, 0);
    k_bnadd<<<(NNODE * CH + T - 1) / T, T, 0, stream>>>(
        t2, h, bn1_g + l * CH, bn1_b + l * CH, bn1_m + l * CH, bn1_v + l * CH, h1b);
    // attention
    gemm(stream, h, nullptr, CH, qkvWt + l * 3 * cc, qkv_b + l * 3 * CH, qkvb, 3 * CH,
         NNODE, 3 * CH, CH, CH, 0);
    k_attn<<<NGRAPH * NHEAD, 256, NPG * NPG * 4 + NPG * HDIM * 4, stream>>>(qkvb, oat);
    gemm(stream, oat, nullptr, CH, oWt + l * cc, o_b + l * CH, ob, CH, NNODE, CH, CH, CH, 0);
    k_bnadd<<<(NNODE * CH + T - 1) / T, T, 0, stream>>>(
        ob, h, bn2_g + l * CH, bn2_b + l * CH, bn2_m + l * CH, bn2_v + l * CH, h2b);
    // combine + FF
    k_add<<<(NNODE * CH + T - 1) / T, T, 0, stream>>>(h1b, h2b, outb, NNODE * CH);
    gemm(stream, outb, nullptr, CH, m1Wt + l * 2 * cc, m1_b + l * 2 * CH, ff1, 2 * CH,
         NNODE, 2 * CH, CH, CH, 1);
    gemm(stream, ff1, nullptr, 2 * CH, m2Wt + l * 2 * cc, m2_b + l * CH, ff2, CH,
         NNODE, CH, 2 * CH, 2 * CH, 0);
    k_bnadd<<<(NNODE * CH + T - 1) / T, T, 0, stream>>>(
        outb, ff2, bn3_g + l * CH, bn3_b + l * CH, bn3_m + l * CH, bn3_v + l * CH, h);
  }
  // ---- pool + head ----
  k_pool<<<NGRAPH, CH, 0, stream>>>(h, pool);
  gemm(stream, pool, nullptr, CH, h1Wt, h1_b, hd1, 64, NGRAPH, 64, CH, CH, 1);
  k_head2<<<1, NGRAPH, 0, stream>>>(hd1, h2_W, h2_b, out);
}
